// SimpleRNN_19061064859747
// MI455X (gfx1250) — matrix operand order repaired and checked by reading, not compiled, not run on hardware
//
#include <hip/hip_runtime.h>
#include <math.h>
#include <stdint.h>

// Problem constants
#define Bb   4
#define Tt   2048
#define Hh   512
#define Kk   128
#define Vv   128
#define Oo   512
#define BT   (Bb * Tt)          // 8192
#define NPROJ (3 * Kk + Vv)     // 512

typedef __bf16 bf16_t;
typedef __attribute__((ext_vector_type(16))) __bf16 v16bf;
typedef __attribute__((ext_vector_type(8)))  float  v8f;

// Fast transcendental helpers: v_exp_f32 + v_rcp_f32 (no IEEE div refinement).
__device__ inline float fast_sigmoid(float x) {
  return __builtin_amdgcn_rcpf(1.f + __expf(-x));
}
__device__ inline float fast_tanh(float x) {
  // tanh(x) = 1 - 2/(exp(2x)+1)
  return 1.f - 2.f * __builtin_amdgcn_rcpf(1.f + __expf(2.f * x));
}

// ---------------------------------------------------------------------------
// CDNA5 async global->LDS copy (GLOBAL_LOAD_ASYNC_TO_LDS_B128, ASYNCcnt).
// GVS addressing: mem = SGPR64 base + VGPR32 byte offset. VDST VGPR carries
// the LDS byte address. Tracked by ASYNCcnt; drained with s_wait_asynccnt.
// ---------------------------------------------------------------------------
__device__ inline uint32_t lds_addr32(const void* p) {
  // Generic LDS pointers carry the LDS byte offset in the low 32 bits
  // (ISA 10.2, LDS aperture).
  return (uint32_t)(uintptr_t)p;
}

__device__ inline void async_load_b128(uint32_t lds_off, const void* base,
                                       int byte_off) {
  asm volatile("global_load_async_to_lds_b128 %0, %1, %2"
               :
               : "v"(lds_off), "v"(byte_off), "s"(base)
               : "memory");
}

__device__ inline void wait_async0() {
  asm volatile("s_wait_asynccnt 0x0" ::: "memory");
}

// ---------------------------------------------------------------------------
// WMMA fragment loaders (wave32); A and B both use k = 16*(lane/16) + i
// ---------------------------------------------------------------------------

// A matrix 16x32 bf16, row-major in LDS with row stride `ld`.
__device__ inline v16bf load_a_frag(const bf16_t* src, int ld) {
  const int lane = threadIdx.x & 31;
  const int row  = lane & 15;
  const int kb   = (lane >> 4) << 4;   // 0 or 16
  v16bf a;
#pragma unroll
  for (int i = 0; i < 16; ++i) {
    a[i]     = src[row * ld + kb + i];
  }
  return a;
}

// B matrix 32x16 bf16 where logical B[k][n] = src[k*ld + n] (KxN row-major).
__device__ inline v16bf load_b_frag_kn(const bf16_t* src, int ld) {
  const int lane = threadIdx.x & 31;
  const int col  = lane & 15;
  const int kb   = (lane >> 4) << 4;   // 0 or 16
  v16bf b;
#pragma unroll
  for (int i = 0; i < 16; ++i) b[i] = src[(kb + i) * ld + col];
  return b;
}

// B matrix 32x16 where logical B[k][n] = src[n*ld + k] (stored N x K).
__device__ inline v16bf load_b_frag_nk(const bf16_t* src, int ld) {
  const int lane = threadIdx.x & 31;
  const int col  = lane & 15;
  const int kb   = (lane >> 4) << 4;   // 0 or 16
  v16bf b;
#pragma unroll
  for (int i = 0; i < 16; ++i) b[i] = src[col * ld + kb + i];
  return b;
}

__device__ inline v8f wmma_bf16(v16bf a, v16bf b, v8f c) {
  return __builtin_amdgcn_wmma_f32_16x16x32_bf16(false, a, false, b,
                                                 (short)0, c, false, false);
}

// ---------------------------------------------------------------------------
// Kernel 1: qkgv = hidden @ proj_w^T + proj_b, split + activations.
// Block 128 thr (4 waves), tile 64(M) x 64(N), K-chunks of 32 over H=512.
// ---------------------------------------------------------------------------
#define GEMM_LD 40   // 32 + 8 bf16 pad; 80B row stride (16B aligned)

__global__ __launch_bounds__(128) void proj_gemm_kernel(
    const float* __restrict__ X,     // [BT][H]
    const float* __restrict__ W,     // [NPROJ][H]
    const float* __restrict__ bias,  // [NPROJ]
    float* __restrict__ q, float* __restrict__ kg,
    float* __restrict__ g, float* __restrict__ v) {
  __shared__ alignas(16) bf16_t Xs[64 * GEMM_LD];
  __shared__ alignas(16) bf16_t Ws[64 * GEMM_LD];
  const int m0 = blockIdx.x * 64;
  const int n0 = blockIdx.y * 64;
  const int tid = threadIdx.x;
  const int wave = tid >> 5;

  v8f acc[4] = {};
  for (int h0 = 0; h0 < Hh; h0 += 32) {
    for (int i = tid; i < 64 * 32; i += 128) {
      const int r = i >> 5, c = i & 31;
      Xs[r * GEMM_LD + c] = (bf16_t)X[(size_t)(m0 + r) * Hh + h0 + c];
      Ws[r * GEMM_LD + c] = (bf16_t)W[(size_t)(n0 + r) * Hh + h0 + c];
    }
    if (h0 + 32 < Hh) {  // prefetch next tiles (global_prefetch_b8)
      __builtin_prefetch(&X[(size_t)(m0 + (tid & 63)) * Hh + h0 + 32], 0, 1);
      __builtin_prefetch(&W[(size_t)(n0 + (tid & 63)) * Hh + h0 + 32], 0, 1);
    }
    __syncthreads();
    const v16bf a = load_a_frag(&Xs[(wave * 16) * GEMM_LD], GEMM_LD);
#pragma unroll
    for (int nt = 0; nt < 4; ++nt) {
      const v16bf b = load_b_frag_nk(&Ws[(nt * 16) * GEMM_LD], GEMM_LD);
      acc[nt] = wmma_bf16(a, b, acc[nt]);
    }
    __syncthreads();
  }

  const int lane = tid & 31;
  const int col16 = lane & 15;
  const int rbase = (lane >> 4) << 3;
#pragma unroll
  for (int nt = 0; nt < 4; ++nt) {
    const int n = n0 + nt * 16 + col16;
    const float bn = bias[n];
    const int region = n >> 7;   // 0:q 1:kgate 2:gate 3:value
    const int cc = n & 127;
#pragma unroll
    for (int r = 0; r < 8; ++r) {
      const int m = m0 + wave * 16 + rbase + r;
      float val = acc[nt][r] + bn;
      float* dst;
      if (region == 0) { dst = q; }
      else if (region == 1) { val = fast_sigmoid(val); dst = kg; }
      else if (region == 2) { val = fast_sigmoid(val); dst = g; }
      else { val = fast_tanh(val); dst = v; }
      dst[(size_t)m * Kk + cc] = val;
    }
  }
}

// ---------------------------------------------------------------------------
// Kernel 2: gated scan. One wave per (batch, v-column); barrier-free.
// Each lane holds C[k] for k = lane + 32*j, j=0..3. ov emitted as bf16
// (all downstream consumers are bf16 WMMA operands).
// ---------------------------------------------------------------------------
__global__ __launch_bounds__(256) void scan_kernel(
    const float* __restrict__ q,  const float* __restrict__ kg,
    const float* __restrict__ g,  const float* __restrict__ v,
    const float* __restrict__ cell,
    bf16_t* __restrict__ ov16, float* __restrict__ cfinal) {
  const int b      = blockIdx.x >> 4;        // 0..3
  const int vchunk = blockIdx.x & 15;        // 0..15
  const int wave = threadIdx.x >> 5;
  const int lane = threadIdx.x & 31;
  const int vcol = vchunk * 8 + wave;        // 0..127

  float C0 = cell[((size_t)b * Kk + lane      ) * Vv + vcol];
  float C1 = cell[((size_t)b * Kk + lane + 32 ) * Vv + vcol];
  float C2 = cell[((size_t)b * Kk + lane + 64 ) * Vv + vcol];
  float C3 = cell[((size_t)b * Kk + lane + 96 ) * Vv + vcol];

  const float* qb_ = q  + (size_t)b * Tt * Kk;
  const float* kb_ = kg + (size_t)b * Tt * Kk;
  const float* gb_ = g  + (size_t)b * Tt * Kk;
  const float* vb_ = v  + (size_t)b * Tt * Vv;
  bf16_t* ovb_ = ov16 + (size_t)b * Tt * Vv;

  for (int t = 0; t < Tt; ++t) {
    const size_t base = (size_t)t * Kk;
    const float vt = vb_[(size_t)t * Vv + vcol];
    C0 = C0 * gb_[base + lane     ] + kb_[base + lane     ] * vt;
    C1 = C1 * gb_[base + lane + 32] + kb_[base + lane + 32] * vt;
    C2 = C2 * gb_[base + lane + 64] + kb_[base + lane + 64] * vt;
    C3 = C3 * gb_[base + lane + 96] + kb_[base + lane + 96] * vt;
    float p = C0 * qb_[base + lane     ] + C1 * qb_[base + lane + 32] +
              C2 * qb_[base + lane + 64] + C3 * qb_[base + lane + 96];
    p += __shfl_xor(p, 16);
    p += __shfl_xor(p, 8);
    p += __shfl_xor(p, 4);
    p += __shfl_xor(p, 2);
    p += __shfl_xor(p, 1);
    if (lane == 0) ovb_[(size_t)t * Vv + vcol] = (bf16_t)p;
  }
  cfinal[((size_t)b * Kk + lane      ) * Vv + vcol] = C0;
  cfinal[((size_t)b * Kk + lane + 32 ) * Vv + vcol] = C1;
  cfinal[((size_t)b * Kk + lane + 64 ) * Vv + vcol] = C2;
  cfinal[((size_t)b * Kk + lane + 96 ) * Vv + vcol] = C3;
}

// ---------------------------------------------------------------------------
// Kernel 3: causal self-attention on ov (flash-style, online softmax).
// Block = 128 thr (4 waves), 64 query rows per block, 16 per wave.
// Q/K tiles stream global->LDS via async B128 copies; K is double-buffered
// so the next tile loads while WMMAs consume the current one.
// ---------------------------------------------------------------------------
#define QK_LD 136   // 128 + 8 bf16 pad; 272B row stride (16B aligned)
#define P_LD  72    // 64  + 8 bf16 pad; 144B row stride (16B aligned)

__global__ __launch_bounds__(128) void attn_kernel(
    const bf16_t* __restrict__ ov16, bf16_t* __restrict__ ctx16) {
  const int b  = blockIdx.y;
  const int qb = blockIdx.x;           // 0..31
  const int m0 = qb * 64;
  const int tid  = threadIdx.x;
  const int wave = tid >> 5;
  const int lane = tid & 31;
  const int col16 = lane & 15;
  const int rbase = (lane >> 4) << 3;

  __shared__ alignas(16) bf16_t Qs[64 * QK_LD];
  __shared__ alignas(16) bf16_t Ksb[2][64 * QK_LD];
  __shared__ alignas(16) bf16_t Ps[4][16 * P_LD];

  const bf16_t* ovb = ov16 + (size_t)b * Tt * Vv;   // row stride 256 bytes

  // Stage Q tile and K tile 0 asynchronously: 64 rows x 16 b128 chunks each.
  for (int c = tid; c < 64 * 16; c += 128) {
    const int row = c >> 4, seg = c & 15;
    const uint32_t loff = (uint32_t)(row * QK_LD * 2 + seg * 16);
    async_load_b128(lds_addr32(Qs) + loff, ovb, (m0 + row) * 256 + seg * 16);
    async_load_b128(lds_addr32(Ksb[0]) + loff, ovb, row * 256 + seg * 16);
  }
  wait_async0();
  __syncthreads();

  v8f oacc[8] = {};
  float mrow[8], lrow[8];
#pragma unroll
  for (int r = 0; r < 8; ++r) { mrow[r] = -3.0e38f; lrow[r] = 0.f; }

  for (int j = 0; j <= qb; ++j) {
    const bf16_t* Ks = Ksb[j & 1];
    if (j < qb) {  // kick off next key tile into the other buffer
      bf16_t* Kn = Ksb[(j + 1) & 1];
      for (int c = tid; c < 64 * 16; c += 128) {
        const int row = c >> 4, seg = c & 15;
        async_load_b128(lds_addr32(Kn) + (uint32_t)(row * QK_LD * 2 + seg * 16),
                        ovb, ((j + 1) * 64 + row) * 256 + seg * 16);
      }
    }

    // S(16x64) = Q(16x128) . K^T
    v8f sarr[4];
#pragma unroll
    for (int nt = 0; nt < 4; ++nt) {
      v8f s = {};
#pragma unroll
      for (int kk = 0; kk < 4; ++kk) {
        const v16bf a  = load_a_frag(&Qs[(wave * 16) * QK_LD + kk * 32], QK_LD);
        const v16bf bb = load_b_frag_nk(&Ks[(nt * 16) * QK_LD + kk * 32], QK_LD);
        s = wmma_bf16(a, bb, s);
      }
      sarr[nt] = s;
    }

    // causal mask + per-row max
    float rmax[8];
#pragma unroll
    for (int r = 0; r < 8; ++r) rmax[r] = -3.0e38f;
#pragma unroll
    for (int nt = 0; nt < 4; ++nt) {
#pragma unroll
      for (int r = 0; r < 8; ++r) {
        const int key = j * 64 + nt * 16 + col16;
        const int row = m0 + wave * 16 + rbase + r;
        float sv = sarr[nt][r];
        if (key > row) sv = -3.0e38f;
        sarr[nt][r] = sv;
        rmax[r] = fmaxf(rmax[r], sv);
      }
    }
#pragma unroll
    for (int r = 0; r < 8; ++r) {     // reduce across 16 lanes of half-wave
      rmax[r] = fmaxf(rmax[r], __shfl_xor(rmax[r], 1));
      rmax[r] = fmaxf(rmax[r], __shfl_xor(rmax[r], 2));
      rmax[r] = fmaxf(rmax[r], __shfl_xor(rmax[r], 4));
      rmax[r] = fmaxf(rmax[r], __shfl_xor(rmax[r], 8));
    }

    float scale[8], rsum[8];
#pragma unroll
    for (int r = 0; r < 8; ++r) {
      const float nm = fmaxf(mrow[r], rmax[r]);
      scale[r] = __expf(mrow[r] - nm);
      mrow[r] = nm;
      rsum[r] = 0.f;
    }
#pragma unroll
    for (int nt = 0; nt < 4; ++nt) {
#pragma unroll
      for (int r = 0; r < 8; ++r) {
        const float p = __expf(sarr[nt][r] - mrow[r]);
        rsum[r] += p;
        Ps[wave][(rbase + r) * P_LD + nt * 16 + col16] = (bf16_t)p;
      }
    }
#pragma unroll
    for (int r = 0; r < 8; ++r) {
      rsum[r] += __shfl_xor(rsum[r], 1);
      rsum[r] += __shfl_xor(rsum[r], 2);
      rsum[r] += __shfl_xor(rsum[r], 4);
      rsum[r] += __shfl_xor(rsum[r], 8);
      lrow[r] = lrow[r] * scale[r] + rsum[r];
    }
#pragma unroll
    for (int dt = 0; dt < 8; ++dt)
#pragma unroll
      for (int r = 0; r < 8; ++r) oacc[dt][r] *= scale[r];

    // ctx(16x128) += P(16x64) . V(64x128)   (V == Ks, row-major)
    // LDS ops are in-order within a wave, so no barrier needed for Ps.
#pragma unroll
    for (int kk = 0; kk < 2; ++kk) {
      const v16bf a = load_a_frag(&Ps[wave][kk * 32], P_LD);
#pragma unroll
      for (int dt = 0; dt < 8; ++dt) {
        const v16bf bb = load_b_frag_kn(&Ks[(kk * 32) * QK_LD + dt * 16], QK_LD);
        oacc[dt] = wmma_bf16(a, bb, oacc[dt]);
      }
    }

    if (j < qb) wait_async0();   // next K tile resident
    __syncthreads();             // all waves done reading / loading
  }

  bf16_t* ctxb = ctx16 + (size_t)b * Tt * Vv;
  float rl[8];
#pragma unroll
  for (int r = 0; r < 8; ++r) rl[r] = __builtin_amdgcn_rcpf(lrow[r]);
#pragma unroll
  for (int dt = 0; dt < 8; ++dt) {
#pragma unroll
    for (int r = 0; r < 8; ++r) {
      const int row = m0 + wave * 16 + rbase + r;
      ctxb[(size_t)row * Vv + dt * 16 + col16] = (bf16_t)(oacc[dt][r] * rl[r]);
    }
  }
}

// ---------------------------------------------------------------------------
// Kernel 4: outputs = concat([ov, ctx]) @ out_w^T + out_b
// X tiles (bf16) stream to LDS via async B128; W tiles convert f32->bf16.
// ---------------------------------------------------------------------------
__global__ __launch_bounds__(128) void out_gemm_kernel(
    const bf16_t* __restrict__ ov16, const bf16_t* __restrict__ ctx16,
    const float* __restrict__ W,   // [O][2V]
    const float* __restrict__ bias, float* __restrict__ out) {
  __shared__ alignas(16) bf16_t Xs[64 * GEMM_LD];
  __shared__ alignas(16) bf16_t Ws[64 * GEMM_LD];
  const int m0 = blockIdx.x * 64;
  const int n0 = blockIdx.y * 64;
  const int tid = threadIdx.x;
  const int wave = tid >> 5;

  v8f acc[4] = {};
  for (int h0 = 0; h0 < 2 * Vv; h0 += 32) {
    // async-stage X tile: 64 rows x 4 b128 chunks (32 bf16 per row)
    const bf16_t* xb = (h0 < Vv) ? ov16 : ctx16;
    const int hb = (h0 < Vv) ? h0 : h0 - Vv;   // column offset within source
    for (int c = tid; c < 64 * 4; c += 128) {
      const int row = c >> 2, seg = c & 3;
      async_load_b128(lds_addr32(Xs) + (uint32_t)(row * GEMM_LD * 2 + seg * 16),
                      xb, (m0 + row) * 256 + hb * 2 + seg * 16);
    }
    for (int i = tid; i < 64 * 32; i += 128) {
      const int r = i >> 5, c = i & 31;
      Ws[r * GEMM_LD + c] = (bf16_t)W[(size_t)(n0 + r) * (2 * Vv) + h0 + c];
    }
    if (h0 + 32 < 2 * Vv)
      __builtin_prefetch(&W[(size_t)(n0 + (tid & 63)) * (2 * Vv) + h0 + 32], 0, 1);
    wait_async0();
    __syncthreads();
    const v16bf a = load_a_frag(&Xs[(wave * 16) * GEMM_LD], GEMM_LD);
#pragma unroll
    for (int nt = 0; nt < 4; ++nt) {
      const v16bf bb = load_b_frag_nk(&Ws[(nt * 16) * GEMM_LD], GEMM_LD);
      acc[nt] = wmma_bf16(a, bb, acc[nt]);
    }
    __syncthreads();
  }

  const int lane = tid & 31;
  const int col16 = lane & 15;
  const int rbase = (lane >> 4) << 3;
#pragma unroll
  for (int nt = 0; nt < 4; ++nt) {
    const int n = n0 + nt * 16 + col16;
    const float bn = bias[n];
#pragma unroll
    for (int r = 0; r < 8; ++r) {
      const int m = m0 + wave * 16 + rbase + r;
      out[(size_t)m * Oo + n] = acc[nt][r] + bn;
    }
  }
}

// ---------------------------------------------------------------------------
extern "C" void kernel_launch(void* const* d_in, const int* in_sizes, int n_in,
                              void* d_out, int out_size, void* d_ws,
                              size_t ws_size, hipStream_t stream) {
  const float* hidden = (const float*)d_in[0];
  const float* cell   = (const float*)d_in[1];
  const float* proj_w = (const float*)d_in[2];
  const float* proj_b = (const float*)d_in[3];
  const float* out_w  = (const float*)d_in[4];
  const float* out_b  = (const float*)d_in[5];
  float* out = (float*)d_out;
  float* ws  = (float*)d_ws;

  const size_t CH = (size_t)BT * Kk;  // 1,048,576 floats per channel buffer
  float* q   = ws;
  float* kg  = ws + CH;
  float* g   = ws + 2 * CH;
  float* v   = ws + 3 * CH;
  bf16_t* ov16  = (bf16_t*)(ws + 4 * CH);          // [BT][V] bf16
  bf16_t* ctx16 = ov16 + (size_t)BT * Vv;          // [BT][V] bf16
  float* cfinal = out + (size_t)BT * Oo;           // C_final follows outputs

  // 1) projection + activations
  proj_gemm_kernel<<<dim3(BT / 64, NPROJ / 64), 128, 0, stream>>>(
      hidden, proj_w, proj_b, q, kg, g, v);
  // 2) gated scan (wave-per-column, barrier-free recurrence) -> bf16 ov
  scan_kernel<<<dim3(Bb * (Vv / 8)), 256, 0, stream>>>(
      q, kg, g, v, cell, ov16, cfinal);
  // 3) causal attention over scan outputs (async double-buffered K tiles)
  attn_kernel<<<dim3(Tt / 64, Bb), 128, 0, stream>>>(ov16, ctx16);
  // 4) output projection (async-staged X tiles)
  out_gemm_kernel<<<dim3(BT / 64, Oo / 64), 128, 0, stream>>>(
      ov16, ctx16, out_w, out_b, out);
}
